// Qwen2StyleMoE_71640054497663
// MI455X (gfx1250) — matrix operand order repaired and checked by reading, not compiled, not run on hardware
//
#include <hip/hip_runtime.h>
#include <hip/hip_bf16.h>
#include <math.h>

// ---------------- problem constants (exact divisors chosen for tiles) -------
#define T_TOK   2048          // B*S
#define H_DIM   2048
#define NEXP    8
#define FD      1408
#define SFD     5632
#define KOFF_SH 11264         // NEXP*FD
#define KTOT    16896         // NEXP*FD + SFD

typedef __bf16 bf16_t;
typedef __attribute__((ext_vector_type(16))) __bf16 v16bf;
typedef __attribute__((ext_vector_type(8)))  __bf16 v8bf;
typedef __attribute__((ext_vector_type(8)))  float  v8f;

#define BK 64
#define LDT 72                // padded LDS row stride (halves): 144B, bank-spread

static __device__ __forceinline__ unsigned short bfbits(float f) {
  unsigned u = __builtin_bit_cast(unsigned, f);
  u += 0x7FFFu + ((u >> 16) & 1u);            // round-to-nearest-even
  return (unsigned short)(u >> 16);
}
static __device__ __forceinline__ bf16_t cvt_bf16(float f) {
  unsigned short s = bfbits(f);
  return __builtin_bit_cast(bf16_t, s);
}
static __device__ __forceinline__ uint4 pack8(float4 a, float4 b) {
  uint4 r;
  r.x = (unsigned)bfbits(a.x) | ((unsigned)bfbits(a.y) << 16);
  r.y = (unsigned)bfbits(a.z) | ((unsigned)bfbits(a.w) << 16);
  r.z = (unsigned)bfbits(b.x) | ((unsigned)bfbits(b.y) << 16);
  r.w = (unsigned)bfbits(b.z) | ((unsigned)bfbits(b.w) << 16);
  return r;
}

// ---------------- kernel 1: fp32 -> bf16 convert of hidden states ----------
__global__ void moe_cvt_x_kernel(const float* __restrict__ x, bf16_t* __restrict__ xb) {
  int i = blockIdx.x * blockDim.x + threadIdx.x;        // one float4 each
  float4 v = ((const float4*)x)[i];
  uint2 o;
  o.x = (unsigned)bfbits(v.x) | ((unsigned)bfbits(v.y) << 16);
  o.y = (unsigned)bfbits(v.z) | ((unsigned)bfbits(v.w) << 16);
  ((uint2*)xb)[i] = o;
}

// ---------------- kernel 2: router (softmax/top-2) + shared sigmoid gate ---
__global__ void moe_router_kernel(const float* __restrict__ x,
                                  const float* __restrict__ gate_w,
                                  const float* __restrict__ se_gate_w,
                                  float* __restrict__ wm,
                                  float* __restrict__ gate_val) {
  const int lane = threadIdx.x & 31;
  const int wave = threadIdx.x >> 5;
  const int t = blockIdx.x * 4 + wave;                  // wave per token
  float acc[9];
#pragma unroll
  for (int i = 0; i < 9; ++i) acc[i] = 0.f;
  const float* xr = x + (size_t)t * H_DIM;
  for (int h = lane; h < H_DIM; h += 32) {
    float xv = xr[h];
#pragma unroll
    for (int e = 0; e < 8; ++e) acc[e] += xv * gate_w[e * H_DIM + h];
    acc[8] += xv * se_gate_w[h];
  }
#pragma unroll
  for (int i = 0; i < 9; ++i)
#pragma unroll
    for (int off = 16; off > 0; off >>= 1)
      acc[i] += __shfl_xor(acc[i], off, 32);
  if (lane == 0) {
    float m = acc[0];
#pragma unroll
    for (int e = 1; e < 8; ++e) m = fmaxf(m, acc[e]);
    float ex[8], sum = 0.f;
#pragma unroll
    for (int e = 0; e < 8; ++e) { ex[e] = __expf(acc[e] - m); sum += ex[e]; }
    int i1 = 0;
#pragma unroll
    for (int e = 1; e < 8; ++e) if (ex[e] > ex[i1]) i1 = e;
    int i2 = (i1 == 0) ? 1 : 0;
#pragma unroll
    for (int e = 0; e < 8; ++e) if (e != i1 && ex[e] > ex[i2]) i2 = e;
    float inv = 1.f / sum;                              // NORM_TOPK_PROB=False
#pragma unroll
    for (int e = 0; e < 8; ++e)
      wm[t * 8 + e] = (e == i1 || e == i2) ? ex[e] * inv : 0.f;
    gate_val[t] = 1.f / (1.f + __expf(-acc[8]));
  }
}

// ---------------- kernel 3: fused gate/up GEMM + silu*u*scale epilogue -----
// Block tile 128x128, 8 waves as 2(M) x 4(N); wave tile 64x32 per matrix.
// Per k32: 4 A-frags + 2+2 B-frags -> 16 WMMAs (1.0 ds-frag-loads per WMMA).
#define G_BM 128
#define G_BN 128
__global__ __launch_bounds__(256)
void moe_gateup_kernel(const bf16_t* __restrict__ xb,
                       const float* __restrict__ Wg,
                       const float* __restrict__ Wu,
                       const float* __restrict__ scale, int scale_stride,
                       unsigned long long w_z_stride,
                       int out_col_base, int z_col_stride,
                       bf16_t* __restrict__ A2) {
  __shared__ __align__(32) bf16_t sA[G_BM][LDT];
  __shared__ __align__(32) bf16_t sG[G_BN][LDT];
  __shared__ __align__(32) bf16_t sU[G_BN][LDT];

  const int tid = threadIdx.x;
  const int lane = tid & 31, wave = tid >> 5;
  const int wmi = wave & 1, wni = wave >> 1;            // 2 x 4 wave grid
  const int hs = lane >> 4, l15 = lane & 15;
  const int z = blockIdx.z;
  const int row0 = blockIdx.x * G_BM;
  const int col0 = blockIdx.y * G_BN;
  const float* wg = Wg + (size_t)z * w_z_stride;
  const float* wu = Wu + (size_t)z * w_z_stride;

  const int arow = tid >> 1, acol = (tid & 1) * 32;     // A staging: 32 halves/thr
  const int brow = tid >> 1, bcol = (tid & 1) * 32;     // B staging: 32 floats/thr

  v8f accg[4][2] = {}; v8f accu[4][2] = {};

  for (int kk = 0; kk < H_DIM; kk += BK) {
    // stage A (already bf16): 4x 16B per thread
    {
      const uint4* src = (const uint4*)(xb + (size_t)(row0 + arow) * H_DIM + kk + acol);
#pragma unroll
      for (int j = 0; j < 4; ++j)
        *(uint4*)&sA[arow][acol + j * 8] = src[j];
    }
    // stage B gate & up: fp32 load, convert to bf16
    {
      const float* g4 = wg + (size_t)(col0 + brow) * H_DIM + kk + bcol;
      const float* u4 = wu + (size_t)(col0 + brow) * H_DIM + kk + bcol;
      if (kk + BK < H_DIM) {                             // -> global_prefetch
        __builtin_prefetch(g4 + BK, 0, 3);
        __builtin_prefetch(u4 + BK, 0, 3);
      }
#pragma unroll
      for (int j = 0; j < 4; ++j) {
        *(uint4*)&sG[brow][bcol + j * 8] =
            pack8(((const float4*)g4)[2 * j], ((const float4*)g4)[2 * j + 1]);
        *(uint4*)&sU[brow][bcol + j * 8] =
            pack8(((const float4*)u4)[2 * j], ((const float4*)u4)[2 * j + 1]);
      }
    }
    __syncthreads();
#pragma unroll
    for (int k32 = 0; k32 < BK; k32 += 32) {
      v16bf a[4];
#pragma unroll
      for (int tm = 0; tm < 4; ++tm) {                  // A frag: lane=M, K contiguous
        union { v16bf v; v8bf p[2]; } t_;
        const int r = wmi * 64 + tm * 16 + l15;
        t_.p[0] = *(const v8bf*)&sA[r][k32 + hs * 16];
        t_.p[1] = *(const v8bf*)&sA[r][k32 + hs * 16 + 8];
        a[tm] = t_.v;
      }
#pragma unroll
      for (int tn = 0; tn < 2; ++tn) {                  // B frag: lane=N, K contiguous
        const int n = wni * 32 + tn * 16 + l15;
        union { v16bf v; v8bf p[2]; } bg_, bu_;
        bg_.p[0] = *(const v8bf*)&sG[n][k32 + hs * 16];
        bg_.p[1] = *(const v8bf*)&sG[n][k32 + hs * 16 + 8];
        bu_.p[0] = *(const v8bf*)&sU[n][k32 + hs * 16];
        bu_.p[1] = *(const v8bf*)&sU[n][k32 + hs * 16 + 8];
#pragma unroll
        for (int tm = 0; tm < 4; ++tm) {
          accg[tm][tn] = __builtin_amdgcn_wmma_f32_16x16x32_bf16(
              false, a[tm], false, bg_.v, (short)0, accg[tm][tn], false, false);
          accu[tm][tn] = __builtin_amdgcn_wmma_f32_16x16x32_bf16(
              false, a[tm], false, bu_.v, (short)0, accu[tm][tn], false, false);
        }
      }
    }
    __syncthreads();
  }
  // epilogue: A2[row, col] = silu(g) * u * scale(row)
  const int cbase = out_col_base + z * z_col_stride + col0 + wni * 32;
#pragma unroll
  for (int tm = 0; tm < 4; ++tm)
#pragma unroll
    for (int tn = 0; tn < 2; ++tn)
#pragma unroll
      for (int r = 0; r < 8; ++r) {
        const int row = row0 + wmi * 64 + tm * 16 + hs * 8 + r;   // C layout
        const float g = accg[tm][tn][r];
        const float u = accu[tm][tn][r];
        const float s = scale[row * scale_stride + z];
        const float val = (g / (1.f + __expf(-g))) * u * s;
        const int col = cbase + tn * 16 + l15;
        A2[(size_t)row * KTOT + col] = cvt_bf16(val);
      }
}

// ---------------- kernel 4: single fused down GEMM over concat-K -----------
// out[t,h] = sum_k A2[t,k] * Bcat[h,k],  Bcat = [down_w(e) rows | shared_down]
// Block tile 64x256, 8 waves as 2(M) x 4(N); wave tile 32x64.
#define D_BM 64
#define D_BN 256
__global__ __launch_bounds__(256)
void moe_down_kernel(const bf16_t* __restrict__ A2,
                     const float* __restrict__ dW,    // [E,H,F]
                     const float* __restrict__ sdW,   // [H,SF]
                     float* __restrict__ out) {
  __shared__ __align__(32) bf16_t sA[D_BM][LDT];
  __shared__ __align__(32) bf16_t sB[D_BN][LDT];

  const int tid = threadIdx.x;
  const int lane = tid & 31, wave = tid >> 5;
  const int wmi = wave & 1, wni = wave >> 1;
  const int hs = lane >> 4, l15 = lane & 15;
  const int row0 = blockIdx.x * D_BM;
  const int col0 = blockIdx.y * D_BN;

  const int arow = tid >> 2, acol = (tid & 3) * 16;     // A staging: 16 halves/thr
  const int brow = tid;                                 // B staging: 64 floats/thr

  v8f acc[2][4] = {};

  for (int kk = 0; kk < KTOT; kk += BK) {
    {
      const uint4* src = (const uint4*)(A2 + (size_t)(row0 + arow) * KTOT + kk + acol);
      *(uint4*)&sA[arow][acol]     = src[0];
      *(uint4*)&sA[arow][acol + 8] = src[1];
    }
    {
      const int h = col0 + brow;
      const float* src;
      if (kk < KOFF_SH) {                               // expert segment (64 | 1408)
        const int e = kk / FD;
        const int f0 = kk - e * FD;
        src = dW + ((size_t)e * H_DIM + h) * FD + f0;
      } else {                                          // shared segment
        src = sdW + (size_t)h * SFD + (kk - KOFF_SH);
      }
      if (kk + BK < KTOT) __builtin_prefetch(src + BK, 0, 3);
#pragma unroll
      for (int j = 0; j < 8; ++j)
        *(uint4*)&sB[brow][j * 8] =
            pack8(((const float4*)src)[2 * j], ((const float4*)src)[2 * j + 1]);
    }
    __syncthreads();
#pragma unroll
    for (int k32 = 0; k32 < BK; k32 += 32) {
      v16bf a[2];
#pragma unroll
      for (int tm = 0; tm < 2; ++tm) {
        union { v16bf v; v8bf p[2]; } t_;
        const int r = wmi * 32 + tm * 16 + l15;
        t_.p[0] = *(const v8bf*)&sA[r][k32 + hs * 16];
        t_.p[1] = *(const v8bf*)&sA[r][k32 + hs * 16 + 8];
        a[tm] = t_.v;
      }
#pragma unroll
      for (int tn = 0; tn < 4; ++tn) {
        const int n = wni * 64 + tn * 16 + l15;
        union { v16bf v; v8bf p[2]; } b_;
        b_.p[0] = *(const v8bf*)&sB[n][k32 + hs * 16];
        b_.p[1] = *(const v8bf*)&sB[n][k32 + hs * 16 + 8];
#pragma unroll
        for (int tm = 0; tm < 2; ++tm)
          acc[tm][tn] = __builtin_amdgcn_wmma_f32_16x16x32_bf16(
              false, a[tm], false, b_.v, (short)0, acc[tm][tn], false, false);
      }
    }
    __syncthreads();
  }
#pragma unroll
  for (int tm = 0; tm < 2; ++tm)
#pragma unroll
    for (int tn = 0; tn < 4; ++tn)
#pragma unroll
      for (int r = 0; r < 8; ++r) {
        const int row = row0 + wmi * 32 + tm * 16 + hs * 8 + r;
        const int col = col0 + wni * 64 + tn * 16 + l15;
        out[(size_t)row * H_DIM + col] = acc[tm][tn][r];
      }
}

// ---------------- host-side orchestration ----------------------------------
extern "C" void kernel_launch(void* const* d_in, const int* in_sizes, int n_in,
                              void* d_out, int out_size, void* d_ws, size_t ws_size,
                              hipStream_t stream) {
  const float* x    = (const float*)d_in[0];   // [2,1024,2048]
  const float* gw   = (const float*)d_in[1];   // [8,2048]
  const float* gpw  = (const float*)d_in[2];   // [8,1408,2048]
  const float* upw  = (const float*)d_in[3];   // [8,1408,2048]
  const float* dpw  = (const float*)d_in[4];   // [8,2048,1408]
  const float* sgw  = (const float*)d_in[5];   // [5632,2048]
  const float* suw  = (const float*)d_in[6];   // [5632,2048]
  const float* sdw  = (const float*)d_in[7];   // [2048,5632]
  const float* segw = (const float*)d_in[8];   // [1,2048]
  float* out = (float*)d_out;

  // workspace layout (all 256B aligned): xb 8MB | wm 64KB | gv 8KB | A2 66MB
  char* ws = (char*)d_ws;
  bf16_t* xb = (bf16_t*)(ws);
  float*  wm = (float*)(ws + 8388608);
  float*  gv = (float*)(ws + 8388608 + 65536);
  bf16_t* A2 = (bf16_t*)(ws + 8388608 + 65536 + 8192);

  moe_cvt_x_kernel<<<4096, 256, 0, stream>>>(x, xb);
  moe_router_kernel<<<512, 128, 0, stream>>>(x, gw, segw, wm, gv);
  // routed experts: grid z = expert, scale = wm[row*8+e], cols at e*FD
  moe_gateup_kernel<<<dim3(16, 11, 8), 256, 0, stream>>>(
      xb, gpw, upw, wm, /*scale_stride=*/8,
      /*w_z_stride=*/(unsigned long long)FD * H_DIM,
      /*out_col_base=*/0, /*z_col_stride=*/FD, A2);
  // shared expert: scale = gate_val[row], cols at KOFF_SH
  moe_gateup_kernel<<<dim3(16, 44, 1), 256, 0, stream>>>(
      xb, sgw, suw, gv, /*scale_stride=*/1,
      /*w_z_stride=*/0ull,
      /*out_col_base=*/KOFF_SH, /*z_col_stride=*/0, A2);
  // single fused down GEMM: out = A2 (2048 x 16896) * Bcat^T -> fp32
  moe_down_kernel<<<dim3(32, 8), 256, 0, stream>>>(A2, dpw, sdw, out);
}
